// SparseDeformableDETRTransformerEncoder_63050119905797
// MI455X (gfx1250) — matrix operand order repaired and checked by reading, not compiled, not run on hardware
//
#include <hip/hip_runtime.h>
#include <hip/hip_bf16.h>

#define NTOK 21760
#define BB   2
#define DD   256
#define NHD  8
#define NLV  4
#define NPT  4
#define KK   2048
#define FFND 1024
#define NLAY 6
#define DHD  32

typedef __attribute__((ext_vector_type(16))) __bf16 bf16x16;
typedef __attribute__((ext_vector_type(8)))  __bf16 bf16x8;
typedef __attribute__((ext_vector_type(8)))  float  f32x8;

// b128 async-copy element type, matching the builtin's expected signature:
// (v4i addrspace(1)*, v4i addrspace(3)*, imm offset, imm cpol)
typedef int v4i __attribute__((vector_size(16)));
typedef __attribute__((address_space(1))) v4i gv4i;   // global
typedef __attribute__((address_space(3))) v4i lv4i;   // LDS

#define HAS_ASYNC_COPY (__has_builtin(__builtin_amdgcn_global_load_async_to_lds_b128))

template<int N>
__device__ __forceinline__ void wait_asynccnt() {
#if __has_builtin(__builtin_amdgcn_s_wait_asynccnt)
  __builtin_amdgcn_s_wait_asynccnt(N);
#else
  asm volatile("s_wait_asynccnt %0" :: "i"(N) : "memory");
#endif
}

// ---------------------------------------------------------------------------
// Tiled bf16 WMMA GEMM:  C(MxN) = A(MxKd) * W(KdxN) + bias, optional ReLU,
// store f32 or bf16. Block tile 64(M) x 128(N), 8 waves (wave32), each wave
// 16x64 via 4 accumulators sharing one A fragment. Kd%32, M%64, N%128 == 0.
// Global->LDS staging via gfx1250 async-to-LDS DMA, double buffered so tile
// t+1 streams in (ASYNCcnt) while tile t runs on the WMMA pipe.
// ---------------------------------------------------------------------------
template<int RELU, int STORE_BF16>
__global__ void __launch_bounds__(256) k_gemm_bf16(
    const __bf16* __restrict__ A, const __bf16* __restrict__ W,
    const float* __restrict__ bias, float* __restrict__ Cf,
    __bf16* __restrict__ Cb, int M, int Kd, int N)
{
  __shared__ __bf16 smA[2][64 * 32];
  __shared__ __bf16 smB[2][32 * 128];
  const int tid  = threadIdx.x;
  const int lane = tid & 31;
  const int wid  = tid >> 5;
  const int wm   = wid >> 1;       // 0..3  -> M sub-tile
  const int wn   = wid & 1;        // 0..1  -> N sub-tile
  const int m0   = blockIdx.y * 64;
  const int n0   = blockIdx.x * 128;

  f32x8 acc[4];
#pragma unroll
  for (int i = 0; i < 4; ++i)
#pragma unroll
    for (int j = 0; j < 8; ++j) acc[i][j] = 0.f;

  const int arow  = tid >> 2,  achk  = tid & 3;    // 64 rows * 4 x 16B chunks
  const int brow0 = tid >> 4,  bchk  = tid & 15;   // 32 rows * 16 x 16B chunks
  const int brow1 = (tid + 256) >> 4;

  const int nT = Kd >> 5;                          // number of 32-wide K tiles

  // per-wave fragment addresses (constant across tiles)
  const int r     = wm * 16 + (lane & 15);
  const int half8 = (lane >> 4) * 8;

#if HAS_ASYNC_COPY
  auto issueTile = [&](int t, int buf) {
    const int kt = t * 32;
    __builtin_amdgcn_global_load_async_to_lds_b128(
        (gv4i*)(A + (size_t)(m0 + arow) * Kd + kt + achk * 8),
        (lv4i*)(&smA[buf][arow * 32 + achk * 8]), 0, 0);
    __builtin_amdgcn_global_load_async_to_lds_b128(
        (gv4i*)(W + (size_t)(kt + brow0) * N + n0 + bchk * 8),
        (lv4i*)(&smB[buf][brow0 * 128 + bchk * 8]), 0, 0);
    __builtin_amdgcn_global_load_async_to_lds_b128(
        (gv4i*)(W + (size_t)(kt + brow1) * N + n0 + bchk * 8),
        (lv4i*)(&smB[buf][brow1 * 128 + bchk * 8]), 0, 0);
  };

  issueTile(0, 0);
  for (int t = 0; t < nT; ++t) {
    const int cur = t & 1;
    if (t + 1 < nT) {
      issueTile(t + 1, cur ^ 1);   // stream next tile while computing this one
      wait_asynccnt<3>();          // in-order: current stage (3 ops) retired
    } else {
      wait_asynccnt<0>();
    }
    __syncthreads();               // all waves' stage data visible in LDS

    bf16x8 alo = *((const bf16x8*)(&smA[cur][r * 32 + half8]));
    bf16x8 ahi = *((const bf16x8*)(&smA[cur][r * 32 + 16 + half8]));
    bf16x16 av = __builtin_shufflevector(alo, ahi,
        0, 1, 2, 3, 4, 5, 6, 7, 8, 9, 10, 11, 12, 13, 14, 15);
#pragma unroll
    for (int i = 0; i < 4; ++i) {
      const int nb = wn * 64 + i * 16;
      bf16x16 bv;
#pragma unroll
      for (int s = 0; s < 16; ++s)
        bv[s] = smB[cur][((s < 8 ? s : s + 8) + half8) * 128 + nb + (lane & 15)];
      acc[i] = __builtin_amdgcn_wmma_f32_16x16x32_bf16(
          false, av, false, bv, (short)0, acc[i], false, false);
    }
    __syncthreads();               // done reading before buffer is re-filled
  }
#else
  for (int t = 0; t < nT; ++t) {
    const int kt = t * 32;
    ((uint4*)(&smA[0][arow * 32]))[achk] =
        ((const uint4*)(A + (size_t)(m0 + arow) * Kd + kt))[achk];
    *((uint4*)(&smB[0][brow0 * 128 + bchk * 8])) =
        *((const uint4*)(W + (size_t)(kt + brow0) * N + n0 + bchk * 8));
    *((uint4*)(&smB[0][brow1 * 128 + bchk * 8])) =
        *((const uint4*)(W + (size_t)(kt + brow1) * N + n0 + bchk * 8));
    if (kt + 32 < Kd) {            // near-scope prefetch of next K tile
      __builtin_prefetch(A + (size_t)(m0 + arow) * Kd + kt + 32, 0, 3);
      __builtin_prefetch(W + (size_t)(kt + 32 + brow0) * N + n0 + bchk * 8, 0, 3);
    }
    __syncthreads();

    bf16x8 alo = *((const bf16x8*)(&smA[0][r * 32 + half8]));
    bf16x8 ahi = *((const bf16x8*)(&smA[0][r * 32 + 16 + half8]));
    bf16x16 av = __builtin_shufflevector(alo, ahi,
        0, 1, 2, 3, 4, 5, 6, 7, 8, 9, 10, 11, 12, 13, 14, 15);
#pragma unroll
    for (int i = 0; i < 4; ++i) {
      const int nb = wn * 64 + i * 16;
      bf16x16 bv;
#pragma unroll
      for (int s = 0; s < 16; ++s)
        bv[s] = smB[0][((s < 8 ? s : s + 8) + half8) * 128 + nb + (lane & 15)];
      acc[i] = __builtin_amdgcn_wmma_f32_16x16x32_bf16(
          false, av, false, bv, (short)0, acc[i], false, false);
    }
    __syncthreads();
  }
#endif

  // C/D layout: VGPR j -> row j + 8*(lane>>4); col = lane&15 within 16-tile.
  const int hi = lane >> 4, nc = lane & 15;
#pragma unroll
  for (int i = 0; i < 4; ++i) {
    const int col = n0 + wn * 64 + i * 16 + nc;
    const float bv = bias ? bias[col] : 0.f;
#pragma unroll
    for (int j = 0; j < 8; ++j) {
      const int row = m0 + wm * 16 + hi * 8 + j;
      float v = acc[i][j] + bv;
      if (RELU) v = v > 0.f ? v : 0.f;
      if (STORE_BF16) Cb[(size_t)row * N + col] = (__bf16)v;
      else            Cf[(size_t)row * N + col] = v;
    }
  }
}

// ---------------------------------------------------------------------------
// Elementwise / gather / sampling kernels
// ---------------------------------------------------------------------------
__global__ void __launch_bounds__(256) k_cvt(const float* __restrict__ s,
                                             __bf16* __restrict__ d, int n) {
  const int t = blockIdx.x * 256 + threadIdx.x;
  if (t < n) d[t] = (__bf16)s[t];
}

__global__ void __launch_bounds__(256) k_init_output(
    const float* __restrict__ value, float* __restrict__ out) {
  const int t = blockIdx.x * 256 + threadIdx.x;      // B*NTOK*DD threads
  const int d = t & 255;
  const int n = (t >> 8) % NTOK;
  const int b = t / (NTOK * 256);
  out[t] = value[((size_t)n * BB + b) * DD + d];
}

__global__ void __launch_bounds__(256) k_gather(
    const float* __restrict__ q, const float* __restrict__ qp,
    const int* __restrict__ topk, float* __restrict__ tgt,
    float* __restrict__ tgtp) {
  const int t = blockIdx.x * 256 + threadIdx.x;      // B*K*D threads
  const int d = t & 255;
  const int k = (t >> 8) & (KK - 1);
  const int b = t >> 19;
  const int n = topk[b * KK + k];
  tgt[t]  = q [((size_t)n * BB + b) * DD + d];
  tgtp[t] = qp[((size_t)n * BB + b) * DD + d];
}

__global__ void __launch_bounds__(256) k_gather_ref(
    const float* __restrict__ rp, const int* __restrict__ topk,
    float* __restrict__ ref) {
  const int t = blockIdx.x * 256 + threadIdx.x;      // B*K*8 threads
  if (t >= BB * KK * 8) return;
  const int c = t & 7;
  const int k = (t >> 3) & (KK - 1);
  const int b = t >> 14;
  const int n = topk[b * KK + k];
  ref[t] = rp[((size_t)b * NTOK + n) * 8 + c];
}

__global__ void __launch_bounds__(256) k_q(const float* __restrict__ a,
                                           const float* __restrict__ b,
                                           __bf16* __restrict__ o) {
  const int t = blockIdx.x * 256 + threadIdx.x;      // B*K*D threads
  o[t] = (__bf16)(a[t] + b[t]);
}

// One lane per (b,k,h,dh); 16 sample points (4 levels x 4 pts), softmax over
// the 16 attention logits per head, align_corners=False bilinear with border
// masking, accumulate in f32, emit bf16 for the W_out GEMM.
__global__ void __launch_bounds__(256) k_sample(
    const float* __restrict__ v, const float* __restrict__ off,
    const float* __restrict__ attnl, const float* __restrict__ ref,
    __bf16* __restrict__ attn_out) {
  const int t  = blockIdx.x * 256 + threadIdx.x;     // B*K*NH*DH threads
  const int dh = t & 31;
  const int h  = (t >> 5) & 7;
  const int k  = (t >> 8) & (KK - 1);
  const int b  = t >> 19;
  const int bk = b * KK + k;

  float e[16];
  float mx = -1e30f;
  const float* lg = attnl + (size_t)bk * 128 + h * 16;
#pragma unroll
  for (int i = 0; i < 16; ++i) { e[i] = lg[i]; mx = fmaxf(mx, e[i]); }
  float den = 0.f;
#pragma unroll
  for (int i = 0; i < 16; ++i) { e[i] = __expf(e[i] - mx); den += e[i]; }
  const float inv = 1.f / den;

  const float* ofp = off + (size_t)bk * 256 + h * 32;
  const float* rfp = ref + (size_t)bk * 8;
  const int HS[4] = {128, 64, 32, 16};               // square levels
  const int ST[4] = {0, 16384, 20480, 21504};
  float out = 0.f;
#pragma unroll
  for (int l = 0; l < 4; ++l) {
    const int Hl = HS[l], Wl = HS[l];
    const float rx = rfp[l * 2], ry = rfp[l * 2 + 1];
    const float* vl = v + ((size_t)b * NTOK + ST[l]) * DD + h * DHD + dh;
#pragma unroll
    for (int p = 0; p < 4; ++p) {
      const float lx = rx * Wl + ofp[l * 8 + p * 2]     - 0.5f;
      const float ly = ry * Hl + ofp[l * 8 + p * 2 + 1] - 0.5f;
      const float x0f = floorf(lx), y0f = floorf(ly);
      const float fx = lx - x0f, fy = ly - y0f;
      const int x0 = (int)x0f, y0 = (int)y0f;
      float s = 0.f;
#pragma unroll
      for (int dy = 0; dy < 2; ++dy)
#pragma unroll
        for (int dx = 0; dx < 2; ++dx) {
          const int xi = x0 + dx, yi = y0 + dy;
          if (xi >= 0 && xi < Wl && yi >= 0 && yi < Hl) {
            const float w = (dx ? fx : 1.f - fx) * (dy ? fy : 1.f - fy);
            s += w * vl[(size_t)(yi * Wl + xi) * DD];
          }
        }
      out += e[l * 4 + p] * inv * s;
    }
  }
  attn_out[(size_t)bk * DD + h * DHD + dh] = (__bf16)out;
}

// Wave-per-row residual-add + LayerNorm over D=256 (8 cols/lane, wave32
// __shfl_xor tree reduce). Optionally also emits a bf16 copy for next GEMM.
__global__ void __launch_bounds__(256) k_add_ln(
    const float* __restrict__ X, const float* __restrict__ Y,
    const float* __restrict__ g, const float* __restrict__ bta,
    float* __restrict__ outf, __bf16* __restrict__ outb, int rows) {
  const int lane = threadIdx.x & 31;
  const int row  = blockIdx.x * 8 + (threadIdx.x >> 5);
  if (row >= rows) return;
  const float* xr = X + (size_t)row * DD;
  const float* yr = Y + (size_t)row * DD;
  float vals[8], s = 0.f, sq = 0.f;
#pragma unroll
  for (int j = 0; j < 8; ++j) {
    const float tv = xr[lane + 32 * j] + yr[lane + 32 * j];
    vals[j] = tv; s += tv; sq += tv * tv;
  }
#pragma unroll
  for (int o = 16; o; o >>= 1) {
    s  += __shfl_xor(s, o);
    sq += __shfl_xor(sq, o);
  }
  const float mu  = s * (1.f / DD);
  const float var = sq * (1.f / DD) - mu * mu;
  const float rs  = rsqrtf(var + 1e-5f);
#pragma unroll
  for (int j = 0; j < 8; ++j) {
    const int c = lane + 32 * j;
    const float o = (vals[j] - mu) * rs * g[c] + bta[c];
    outf[(size_t)row * DD + c] = o;
    if (outb) outb[(size_t)row * DD + c] = (__bf16)o;
  }
}

__global__ void __launch_bounds__(256) k_scatter(
    const float* __restrict__ tgt, const int* __restrict__ topk,
    const int* __restrict__ stn, float* __restrict__ out) {
  const int t = blockIdx.x * 256 + threadIdx.x;      // B*K*D threads
  const int d = t & 255;
  const int k = (t >> 8) & (KK - 1);
  const int b = t >> 19;
  if (k < stn[b])
    out[((size_t)b * NTOK + topk[b * KK + k]) * DD + d] = tgt[t];
}

__global__ void __launch_bounds__(256) k_final(
    const float* __restrict__ out, float* __restrict__ dst) {
  const int t = blockIdx.x * 256 + threadIdx.x;      // B*NTOK*DD threads
  const int d = t & 255;
  const int n = (t >> 8) % NTOK;
  const int b = t / (NTOK * 256);
  dst[((size_t)n * BB + b) * DD + d] = out[t];
}

// ---------------------------------------------------------------------------
// Host launcher
// ---------------------------------------------------------------------------
extern "C" void kernel_launch(void* const* d_in, const int* in_sizes, int n_in,
                              void* d_out, int out_size, void* d_ws, size_t ws_size,
                              hipStream_t stream) {
  (void)in_sizes; (void)n_in; (void)out_size; (void)ws_size;
  const float* query  = (const float*)d_in[0];
  const float* qpos   = (const float*)d_in[1];
  const float* value  = (const float*)d_in[2];
  const float* refp   = (const float*)d_in[3];
  const int*   topk   = (const int*)d_in[4];
  const int*   stn    = (const int*)d_in[5];
  const float* W_off  = (const float*)d_in[6];
  const float* b_off  = (const float*)d_in[7];
  const float* W_attn = (const float*)d_in[8];
  const float* b_attn = (const float*)d_in[9];
  const float* W_v    = (const float*)d_in[10];
  const float* b_v    = (const float*)d_in[11];
  const float* W_out  = (const float*)d_in[12];
  const float* b_out  = (const float*)d_in[13];
  const float* ln1_g  = (const float*)d_in[14];
  const float* ln1_b  = (const float*)d_in[15];
  const float* W1     = (const float*)d_in[16];
  const float* b1     = (const float*)d_in[17];
  const float* W2     = (const float*)d_in[18];
  const float* b2     = (const float*)d_in[19];
  const float* ln2_g  = (const float*)d_in[20];
  const float* ln2_b  = (const float*)d_in[21];

  char* p = (char*)d_ws;
  auto carve = [&](size_t bytes) -> char* {
    char* r = p; p += (bytes + 255) & ~(size_t)255; return r;
  };
  const size_t BND = (size_t)BB * NTOK * DD;         // 11,141,120
  const size_t BKD = (size_t)BB * KK * DD;           //  1,048,576
  float*  outbuf  = (float*)carve(BND * 4);
  __bf16* out_bf  = (__bf16*)carve(BND * 2);
  float*  vbuf    = (float*)carve(BND * 4);
  float*  tgt     = (float*)carve(BKD * 4);
  float*  tgtp    = (float*)carve(BKD * 4);
  float*  refg    = (float*)carve((size_t)BB * KK * 8 * 4);
  __bf16* q_bf    = (__bf16*)carve(BKD * 2);
  float*  offb    = (float*)carve(BKD * 4);
  float*  attnl   = (float*)carve((size_t)BB * KK * 128 * 4);
  __bf16* ao_bf   = (__bf16*)carve(BKD * 2);
  float*  aproj   = (float*)carve(BKD * 4);
  float*  xbuf    = (float*)carve(BKD * 4);
  __bf16* x_bf    = (__bf16*)carve(BKD * 2);
  __bf16* h_bf    = (__bf16*)carve((size_t)BB * KK * FFND * 2);
  float*  ffnb    = (float*)carve(BKD * 4);
  __bf16* Wv_bf   = (__bf16*)carve((size_t)NLAY * DD * DD * 2);
  __bf16* Woff_bf = (__bf16*)carve((size_t)NLAY * DD * DD * 2);
  __bf16* Wat_bf  = (__bf16*)carve((size_t)NLAY * DD * 128 * 2);
  __bf16* Wout_bf = (__bf16*)carve((size_t)NLAY * DD * DD * 2);
  __bf16* W1_bf   = (__bf16*)carve((size_t)NLAY * DD * FFND * 2);
  __bf16* W2_bf   = (__bf16*)carve((size_t)NLAY * FFND * DD * 2);

  auto cvt = [&](const float* s, __bf16* d, size_t n) {
    k_cvt<<<dim3((unsigned)((n + 255) / 256)), dim3(256), 0, stream>>>(s, d, (int)n);
  };
  cvt(W_v,    Wv_bf,   (size_t)NLAY * DD * DD);
  cvt(W_off,  Woff_bf, (size_t)NLAY * DD * DD);
  cvt(W_attn, Wat_bf,  (size_t)NLAY * DD * 128);
  cvt(W_out,  Wout_bf, (size_t)NLAY * DD * DD);
  cvt(W1,     W1_bf,   (size_t)NLAY * DD * FFND);
  cvt(W2,     W2_bf,   (size_t)NLAY * FFND * DD);

  k_init_output<<<dim3((unsigned)(BND / 256)), dim3(256), 0, stream>>>(value, outbuf);
  k_gather<<<dim3((unsigned)(BKD / 256)), dim3(256), 0, stream>>>(query, qpos, topk, tgt, tgtp);
  k_gather_ref<<<dim3((BB * KK * 8) / 256), dim3(256), 0, stream>>>(refp, topk, refg);

  const dim3 blk(256);
  for (int L = 0; L < NLAY; ++L) {
    const __bf16* wv = Wv_bf   + (size_t)L * DD * DD;
    const __bf16* wo = Woff_bf + (size_t)L * DD * DD;
    const __bf16* wa = Wat_bf  + (size_t)L * DD * 128;
    const __bf16* wu = Wout_bf + (size_t)L * DD * DD;
    const __bf16* w1 = W1_bf   + (size_t)L * DD * FFND;
    const __bf16* w2 = W2_bf   + (size_t)L * FFND * DD;

    // v = output @ W_v + b_v   (43520 x 256 x 256)
    k_cvt<<<dim3((unsigned)(BND / 256)), blk, 0, stream>>>(outbuf, out_bf, (int)BND);
    k_gemm_bf16<0, 0><<<dim3(2, (BB * NTOK) / 64), blk, 0, stream>>>(
        out_bf, wv, b_v + L * DD, vbuf, nullptr, BB * NTOK, DD, DD);

    // q = tgt + tgt_pos; offsets + attention logits (4096 rows)
    k_q<<<dim3((unsigned)(BKD / 256)), blk, 0, stream>>>(tgt, tgtp, q_bf);
    k_gemm_bf16<0, 0><<<dim3(2, (BB * KK) / 64), blk, 0, stream>>>(
        q_bf, wo, b_off + L * DD, offb, nullptr, BB * KK, DD, DD);
    k_gemm_bf16<0, 0><<<dim3(1, (BB * KK) / 64), blk, 0, stream>>>(
        q_bf, wa, b_attn + L * 128, attnl, nullptr, BB * KK, DD, 128);

    // deformable sampling -> attn_out (bf16)
    k_sample<<<dim3((BB * KK * NHD * DHD) / 256), blk, 0, stream>>>(
        vbuf, offb, attnl, refg, ao_bf);

    // output projection, residual + LN1
    k_gemm_bf16<0, 0><<<dim3(2, (BB * KK) / 64), blk, 0, stream>>>(
        ao_bf, wu, b_out + L * DD, aproj, nullptr, BB * KK, DD, DD);
    k_add_ln<<<dim3((BB * KK) / 8), blk, 0, stream>>>(
        tgt, aproj, ln1_g + L * DD, ln1_b + L * DD, xbuf, x_bf, BB * KK);

    // FFN: relu(x@W1+b1)@W2+b2, residual + LN2 -> new tgt
    k_gemm_bf16<1, 1><<<dim3(FFND / 128, (BB * KK) / 64), blk, 0, stream>>>(
        x_bf, w1, b1 + L * FFND, nullptr, h_bf, BB * KK, DD, FFND);
    k_gemm_bf16<0, 0><<<dim3(2, (BB * KK) / 64), blk, 0, stream>>>(
        h_bf, w2, b2 + L * DD, ffnb, nullptr, BB * KK, FFND, DD);
    k_add_ln<<<dim3((BB * KK) / 8), blk, 0, stream>>>(
        xbuf, ffnb, ln2_g + L * DD, ln2_b + L * DD, tgt, nullptr, BB * KK);

    // ragged scatter-back into the dense token map
    k_scatter<<<dim3((unsigned)(BKD / 256)), blk, 0, stream>>>(tgt, topk, stn, outbuf);
  }

  k_final<<<dim3((unsigned)(BND / 256)), blk, 0, stream>>>(outbuf, (float*)d_out);
}
